// Model_66254165508717
// MI455X (gfx1250) — matrix operand order repaired and checked by reading, not compiled, not run on hardware
//
#include <hip/hip_runtime.h>

typedef __bf16 bf16;
typedef __attribute__((ext_vector_type(16))) __bf16 v16bf;
typedef __attribute__((ext_vector_type(8)))  float  v8f;

#define WMMA_BF16(a, b, c) \
  __builtin_amdgcn_wmma_f32_16x16x32_bf16(false, (a), false, (b), (short)0, (c), false, false)

// ---------------------------------------------------------------------------
// Model dimensions
// ---------------------------------------------------------------------------
#define BB 8
#define CC 32
#define SS 256          /* B*C series            */
#define LL 336          /* sequence length       */
#define DD 64           /* d_model               */
#define HH 4            /* heads                 */
#define HD 16           /* head dim              */
#define DFF 2048
#define NP 42           /* patches               */
#define PATCH 16
#define STRIDE 8
#define SHARED 5440
#define PL 96
#define ROWS ((long)SS * LL)   /* 86016 token rows */

// ---------------------------------------------------------------------------
// WMMA fragment loaders (bf16 16x16x32, layouts per CDNA5 ISA 7.12.2)
// A row-major [M][K]: lane l holds row m=l&15; elems 0..7 -> K=kb+h*8..+8,
// elems 8..15 -> K=kb+16+h*8..+8 (h = l>>4).
// B pre-transposed [N][K]: lane l holds col n=l&15; same K order as A.
// ---------------------------------------------------------------------------
static __device__ __forceinline__ v16bf load_a_frag(const bf16* A, long lda,
                                                    int mtile, int kb, int lane) {
  int m = lane & 15, h = lane >> 4;
  const bf16* p = A + (long)(mtile * 16 + m) * lda + kb + h * 8;
  v16bf a;
#pragma unroll
  for (int i = 0; i < 8; ++i) { a[i] = p[i]; a[8 + i] = p[16 + i]; }
  return a;
}

static __device__ __forceinline__ v16bf load_b_frag(const bf16* BT, long ldb,
                                                    int ntile, int kb, int lane) {
  int n = lane & 15, h = lane >> 4;
  const bf16* p = BT + (long)(ntile * 16 + n) * ldb + kb + h * 8;
  v16bf b;
#pragma unroll
  for (int i = 0; i < 8; ++i) { b[i] = p[i]; b[8 + i] = p[16 + i]; }
  return b;
}

// ---------------------------------------------------------------------------
// Generic WMMA GEMM (one 16x16 tile per wave) — used only for odd shapes.
// ---------------------------------------------------------------------------
__global__ __launch_bounds__(128) void gemm_wmma(
    const bf16* __restrict__ A, const bf16* __restrict__ BT,
    const float* __restrict__ bias, float* __restrict__ outF,
    bf16* __restrict__ outB, int M, int N, int K, long lda, int relu,
    int total_waves) {
  int wave = (int)((blockIdx.x * (long)blockDim.x + threadIdx.x) >> 5);
  int lane = threadIdx.x & 31;
  if (wave >= total_waves) return;
  int Nt = N >> 4;
  int mt = wave / Nt, nt = wave % Nt;
  v8f acc = {0.f, 0.f, 0.f, 0.f, 0.f, 0.f, 0.f, 0.f};
  for (int kb = 0; kb < K; kb += 32) {
    v16bf a = load_a_frag(A, lda, mt, kb, lane);
    v16bf b = load_b_frag(BT, (long)K, nt, kb, lane);
    acc = WMMA_BF16(a, b, acc);
  }
  int n = nt * 16 + (lane & 15);
  int h = lane >> 4;
  float bv = bias ? bias[n] : 0.f;
#pragma unroll
  for (int r = 0; r < 8; ++r) {
    long m = (long)mt * 16 + r + 8 * h;
    float v = acc[r] + bv;
    if (relu) v = fmaxf(v, 0.f);
    if (outF) outF[m * N + n] = v;
    if (outB) outB[m * N + n] = (bf16)v;
  }
}

// ---------------------------------------------------------------------------
// Register-blocked WMMA GEMM: each wave computes a 32x64 block (2 M-tiles x
// 4 N-tiles = 8 accumulators). A fragments reused 4x, B fragments 2x ->
// ~2.7x less operand traffic per v_wmma. Requires M%32==0, N%64==0.
// Prefetches the next K-chunk (global_prefetch_b8).
// ---------------------------------------------------------------------------
__global__ __launch_bounds__(128) void gemm_wmma_blk(
    const bf16* __restrict__ A, const bf16* __restrict__ BT,
    const float* __restrict__ bias, float* __restrict__ outF,
    bf16* __restrict__ outB, int M, int N, int K, long lda, int relu,
    int total_waves) {
  int wave = (int)((blockIdx.x * (long)blockDim.x + threadIdx.x) >> 5);
  int lane = threadIdx.x & 31;
  if (wave >= total_waves) return;
  int Ns = N >> 6;                       // 64-wide strips
  int ms = wave / Ns, ns = wave % Ns;
  v8f acc[2][4];
#pragma unroll
  for (int mi = 0; mi < 2; ++mi)
#pragma unroll
    for (int j = 0; j < 4; ++j)
#pragma unroll
      for (int r = 0; r < 8; ++r) acc[mi][j][r] = 0.f;

  for (int kb = 0; kb < K; kb += 32) {
    if (kb + 32 < K) {
      __builtin_prefetch(A + (long)(ms * 32 + (lane & 31)) * lda + kb + 32, 0, 1);
      __builtin_prefetch(BT + (long)(ns * 64 + (lane & 31)) * K + kb + 32, 0, 1);
    }
    v16bf a0 = load_a_frag(A, lda, ms * 2 + 0, kb, lane);
    v16bf a1 = load_a_frag(A, lda, ms * 2 + 1, kb, lane);
#pragma unroll
    for (int j = 0; j < 4; ++j) {
      v16bf b = load_b_frag(BT, (long)K, ns * 4 + j, kb, lane);
      acc[0][j] = WMMA_BF16(a0, b, acc[0][j]);
      acc[1][j] = WMMA_BF16(a1, b, acc[1][j]);
    }
  }

  int nlo = lane & 15, h = lane >> 4;
#pragma unroll
  for (int mi = 0; mi < 2; ++mi) {
#pragma unroll
    for (int j = 0; j < 4; ++j) {
      int n = ns * 64 + j * 16 + nlo;
      float bv = bias ? bias[n] : 0.f;
#pragma unroll
      for (int r = 0; r < 8; ++r) {
        long m = (long)(ms * 2 + mi) * 16 + r + 8 * h;
        float v = acc[mi][j][r] + bv;
        if (relu) v = fmaxf(v, 0.f);
        if (outF) outF[m * N + n] = v;
        if (outB) outB[m * N + n] = (bf16)v;
      }
    }
  }
}

// ---------------------------------------------------------------------------
// Weight prep: fp32 [K][N] row-major -> bf16 [N][Kpad] (zero-padded K)
// ---------------------------------------------------------------------------
__global__ void transpose_to_bf16(const float* __restrict__ W,
                                  bf16* __restrict__ WT, int K, int N, int Kpad) {
  long i = blockIdx.x * (long)blockDim.x + threadIdx.x;
  if (i >= (long)N * Kpad) return;
  int kp = (int)(i % Kpad);
  long n = i / Kpad;
  WT[i] = (kp < K) ? (bf16)W[(long)kp * N + n] : (bf16)0.f;
}

// ---------------------------------------------------------------------------
// Latent embedding: h[s,t,d] = x[b,t,c]*latW[d] + latB[d]  (f32 + bf16 copies)
// ---------------------------------------------------------------------------
__global__ void embed_latent(const float* __restrict__ x,
                             const float* __restrict__ lw,
                             const float* __restrict__ lb,
                             float* __restrict__ h, bf16* __restrict__ hB) {
  long i = blockIdx.x * (long)blockDim.x + threadIdx.x;
  if (i >= ROWS * DD) return;
  int d = (int)(i & 63);
  long row = i >> 6;                     // row = s*L + t
  long s = row / LL, t = row % LL;
  int b = (int)(s >> 5), c = (int)(s & 31);
  float v = x[((long)b * LL + t) * CC + c] * lw[d] + lb[d];
  h[i] = v;
  hB[i] = (bf16)v;
}

// ---------------------------------------------------------------------------
// Fused attention: one wave per (series, head, 16-query tile).
// Scores in LDS (f32), shuffle softmax, WMMA against V. Writes bf16 out.
// ---------------------------------------------------------------------------
__global__ __launch_bounds__(64) void attn_fused(const bf16* __restrict__ qB,
                                                 const bf16* __restrict__ kB,
                                                 const bf16* __restrict__ vB,
                                                 bf16* __restrict__ aoB) {
  __shared__ float sc[2][16][LL + 8];
  int wid = threadIdx.x >> 5, lane = threadIdx.x & 31;
  long gw = ((blockIdx.x * 64L + threadIdx.x) >> 5);
  int qt = (int)(gw % 21);
  long sh = gw / 21;
  int head = (int)(sh & 3);
  long s = sh >> 2;
  int m = lane & 15, h = lane >> 4;
  float (*S)[LL + 8] = sc[wid];

  // Q fragment once: head-dim 16 zero-padded to K=32
  v16bf aq;
  const bf16* qr = qB + ((s * LL + qt * 16 + m) * DD + head * HD + h * 8);
#pragma unroll
  for (int i = 0; i < 8; ++i) { aq[i] = qr[i]; aq[8 + i] = (bf16)0.f; }

  for (int nt = 0; nt < 21; ++nt) {
    v16bf bk;
#pragma unroll
    for (int j = 0; j < 16; ++j) bk[j] = (bf16)0.f;
    {
      const bf16* kr = kB + ((s * LL + nt * 16 + (lane & 15)) * DD + head * HD + h * 8);
#pragma unroll
      for (int j = 0; j < 8; ++j) bk[j] = kr[j];
    }
    v8f c = {0.f, 0.f, 0.f, 0.f, 0.f, 0.f, 0.f, 0.f};
    c = WMMA_BF16(aq, bk, c);
#pragma unroll
    for (int r = 0; r < 8; ++r) S[r + 8 * h][nt * 16 + (lane & 15)] = c[r] * 0.25f;
  }

  // softmax over 336 keys: lane pair (m, m+16) splits the row
  float mx = -1e30f;
  for (int j = 0; j < 168; ++j) mx = fmaxf(mx, S[m][h * 168 + j]);
  mx = fmaxf(mx, __shfl_xor(mx, 16, 32));
  float sum = 0.f;
  for (int j = 0; j < 168; ++j) {
    float e = __expf(S[m][h * 168 + j] - mx);
    S[m][h * 168 + j] = e;
    sum += e;
  }
  sum += __shfl_xor(sum, 16, 32);
  float inv = 1.f / sum;

  // att @ V : K = 336 padded to 352
  v8f o = {0.f, 0.f, 0.f, 0.f, 0.f, 0.f, 0.f, 0.f};
  for (int kb = 0; kb < 352; kb += 32) {
    v16bf a;
#pragma unroll
    for (int i = 0; i < 8; ++i) {
      int k0 = kb + h * 8 + i;
      int k1 = kb + 16 + h * 8 + i;
      a[i]     = (k0 < LL) ? (bf16)(S[m][k0] * inv) : (bf16)0.f;
      a[8 + i] = (k1 < LL) ? (bf16)(S[m][k1] * inv) : (bf16)0.f;
    }
    v16bf b;
#pragma unroll
    for (int j = 0; j < 16; ++j) {
      int key = kb + (j < 8 ? h * 8 + j : 16 + h * 8 + (j - 8));
      b[j] = (key < LL) ? vB[(s * LL + key) * DD + head * HD + (lane & 15)]
                        : (bf16)0.f;
    }
    o = WMMA_BF16(a, b, o);
  }
#pragma unroll
  for (int r = 0; r < 8; ++r)
    aoB[(s * LL + qt * 16 + r + 8 * h) * DD + head * HD + (lane & 15)] = (bf16)o[r];
}

// ---------------------------------------------------------------------------
// Residual add + LayerNorm; one wave per row (lane owns dims l and l+32).
// Writes f32 and bf16 copies.
// ---------------------------------------------------------------------------
__global__ __launch_bounds__(256) void residual_ln(
    const float* __restrict__ hIn, const float* __restrict__ fIn,
    const float* __restrict__ g, const float* __restrict__ bta,
    float* __restrict__ hOut, bf16* __restrict__ hOutB, long rows) {
  long row = (blockIdx.x * (long)blockDim.x + threadIdx.x) >> 5;
  int lane = threadIdx.x & 31;
  if (row >= rows) return;
  const float* a = hIn + row * DD;
  const float* f = fIn + row * DD;
  float v0 = a[lane] + f[lane];
  float v1 = a[lane + 32] + f[lane + 32];
  float sum = v0 + v1;
#pragma unroll
  for (int o = 16; o > 0; o >>= 1) sum += __shfl_xor(sum, o, 32);
  float mu = sum * (1.f / 64.f);
  float d0 = v0 - mu, d1 = v1 - mu;
  float vs = d0 * d0 + d1 * d1;
#pragma unroll
  for (int o = 16; o > 0; o >>= 1) vs += __shfl_xor(vs, o, 32);
  float rstd = rsqrtf(vs * (1.f / 64.f) + 1e-5f);
  float o0 = d0 * rstd * g[lane] + bta[lane];
  float o1 = d1 * rstd * g[lane + 32] + bta[lane + 32];
  hOut[row * DD + lane] = o0;
  hOut[row * DD + lane + 32] = o1;
  hOutB[row * DD + lane] = (bf16)o0;
  hOutB[row * DD + lane + 32] = (bf16)o1;
}

// ---------------------------------------------------------------------------
// Fused FFN: relu(h@W1+b1)@W2+b2. Block = 4 waves per 16-row tile.
// Hidden processed in halves of 1024 via LDS (bf16) to avoid HBM round-trip.
// ---------------------------------------------------------------------------
__global__ __launch_bounds__(128) void ffn_fused(
    const bf16* __restrict__ hB, const bf16* __restrict__ W1T,
    const float* __restrict__ b1, const bf16* __restrict__ W2T,
    const float* __restrict__ b2, float* __restrict__ outF) {
  __shared__ bf16 hid[16][1040];
  int mt = blockIdx.x;
  int wid = threadIdx.x >> 5, lane = threadIdx.x & 31;
  int nn = lane & 15, h = lane >> 4;

  v16bf a0 = load_a_frag(hB, DD, mt, 0, lane);
  v16bf a1 = load_a_frag(hB, DD, mt, 32, lane);
  v8f acc2 = {0.f, 0.f, 0.f, 0.f, 0.f, 0.f, 0.f, 0.f};

  for (int half = 0; half < 2; ++half) {
    // stage 1: hidden tile [16 x 1024] -> LDS (relu, bf16)
    for (int nt1 = wid; nt1 < 64; nt1 += 4) {
      int gnt = half * 64 + nt1;
      v8f acc = {0.f, 0.f, 0.f, 0.f, 0.f, 0.f, 0.f, 0.f};
      v16bf b0 = load_b_frag(W1T, DD, gnt, 0, lane);
      v16bf b1f = load_b_frag(W1T, DD, gnt, 32, lane);
      acc = WMMA_BF16(a0, b0, acc);
      acc = WMMA_BF16(a1, b1f, acc);
      float bb = b1[gnt * 16 + nn];
#pragma unroll
      for (int r = 0; r < 8; ++r)
        hid[r + 8 * h][nt1 * 16 + nn] = (bf16)fmaxf(acc[r] + bb, 0.f);
    }
    __syncthreads();
    // stage 2: accumulate hidden-half @ W2  (this wave owns out ntile = wid)
    for (int kb = 0; kb < 1024; kb += 32) {
      v16bf a;
      int m = lane & 15;
#pragma unroll
      for (int i = 0; i < 8; ++i) {
        a[i] = hid[m][kb + h * 8 + i];
        a[8 + i] = hid[m][kb + 16 + h * 8 + i];
      }
      v16bf b = load_b_frag(W2T, DFF, wid, half * 1024 + kb, lane);
      acc2 = WMMA_BF16(a, b, acc2);
    }
    __syncthreads();
  }
  float bb = b2[wid * 16 + nn];
#pragma unroll
  for (int r = 0; r < 8; ++r)
    outF[((long)mt * 16 + r + 8 * h) * DD + wid * 16 + nn] = acc2[r] + bb;
}

// ---------------------------------------------------------------------------
// latent = mean over t of h; written (bf16) into both comb halves' tail cols
// ---------------------------------------------------------------------------
__global__ void latent_mean(const float* __restrict__ h, bf16* __restrict__ comb) {
  int i = blockIdx.x * blockDim.x + threadIdx.x;
  if (i >= SS * DD) return;
  int s = i >> 6, d = i & 63;
  float sum = 0.f;
  for (int t = 0; t < LL; ++t) sum += h[((long)s * LL + t) * DD + d];
  bf16 m = (bf16)(sum / 336.f);
  comb[(long)s * SHARED + NP * 128 + d] = m;
  comb[(long)(SS + s) * SHARED + NP * 128 + d] = m;
}

// ---------------------------------------------------------------------------
// Kalman filter + RTS smoother. Gains are data-independent -> LDS once.
// 256 threads, one per series. trend/resid laid out [B,L,C] like x.
// ---------------------------------------------------------------------------
__global__ __launch_bounds__(256) void kalman_smooth(
    const float* __restrict__ x, float* __restrict__ trend,
    float* __restrict__ resid, float* __restrict__ mf) {
  __shared__ float Kt[LL], G[LL];
  int s = threadIdx.x;
  if (s == 0) {
    float P = 0.5f;          // after t=0 update (P0p=1, R=1 -> K0=0.5)
    Kt[0] = 0.5f;
    G[0] = P / (P + 1.f);
    for (int t = 1; t < LL; ++t) {
      float Pp = P + 1.f;    // Q = 1
      float K = Pp / (Pp + 1.f);
      P = (1.f - K) * Pp;
      Kt[t] = K;
      G[t] = P / (P + 1.f);
    }
  }
  __syncthreads();
  int b = s >> 5, c = s & 31;
  // forward filter
  float m = 0.5f * x[((long)b * LL) * CC + c];
  mf[0 * SS + s] = m;
  for (int t = 1; t < LL; ++t) {
    float y = x[((long)b * LL + t) * CC + c];
    m = m + Kt[t] * (y - m);
    mf[t * SS + s] = m;
  }
  // backward smoother (mp_n == m_t since F=1)
  float ms = mf[(LL - 1) * SS + s];
  {
    long idx = ((long)b * LL + (LL - 1)) * CC + c;
    trend[idx] = ms;
    resid[idx] = x[idx] - ms;
  }
  for (int t = LL - 2; t >= 0; --t) {
    float mt = mf[t * SS + s];
    ms = mt + G[t] * (ms - mt);
    long idx = ((long)b * LL + t) * CC + c;
    trend[idx] = ms;
    resid[idx] = x[idx] - ms;
  }
}

// ---------------------------------------------------------------------------
// Patch embedding via WMMA: relu(patches[16] @ peW[16,128] + peb) -> comb bf16
// One wave per (branch, 16-row tile of (s,patch), 16-col tile). K padded to 32.
// ---------------------------------------------------------------------------
__global__ __launch_bounds__(128) void patch_embed(
    const float* __restrict__ trend, const float* __restrict__ resid,
    const bf16* __restrict__ peWT, const float* __restrict__ peb,
    bf16* __restrict__ comb) {
  long gw = (blockIdx.x * (long)blockDim.x + threadIdx.x) >> 5;
  int lane = threadIdx.x & 31;
  int nt = (int)(gw % 8);
  long t1 = gw / 8;
  int mt = (int)(t1 % 672);
  int branch = (int)(t1 / 672);
  const float* sig = branch ? resid : trend;
  int h = lane >> 4;
  int mrow = mt * 16 + (lane & 15);
  int s = mrow / NP, p = mrow % NP;
  int b = s >> 5, c = s & 31;
  v16bf a;
#pragma unroll
  for (int i = 0; i < 8; ++i) {
    int k = h * 8 + i;                       // element within patch (0..15)
    int t = p * STRIDE + k;
    int tt = t < LL ? t : (LL - 1);          // 'repeat last' padding
    a[i] = (bf16)sig[((long)b * LL + tt) * CC + c];
    a[8 + i] = (bf16)0.f;                    // K padded 16 -> 32
  }
  v16bf bfrag = load_b_frag(peWT, 32, nt, 0, lane);
  v8f acc = {0.f, 0.f, 0.f, 0.f, 0.f, 0.f, 0.f, 0.f};
  acc = WMMA_BF16(a, bfrag, acc);
  int n = nt * 16 + (lane & 15);
  float bb = peb[n];
#pragma unroll
  for (int r = 0; r < 8; ++r) {
    int m2 = mt * 16 + r + 8 * h;
    int s2 = m2 / NP, p2 = m2 % NP;
    float v = fmaxf(acc[r] + bb, 0.f);
    comb[(long)(branch * SS + s2) * SHARED + p2 * 128 + n] = (bf16)v;
  }
}

// ---------------------------------------------------------------------------
// out[b, t, c] = tp[s, t] + rp[s, t]
// ---------------------------------------------------------------------------
__global__ void final_out(const float* __restrict__ tp,
                          const float* __restrict__ rp, float* __restrict__ out) {
  int i = blockIdx.x * blockDim.x + threadIdx.x;
  if (i >= SS * PL) return;
  int s = i / PL, t = i % PL;
  int b = s >> 5, c = s & 31;
  out[((long)b * PL + t) * CC + c] = tp[i] + rp[i];
}

// ===========================================================================
// Host orchestration
// ===========================================================================
extern "C" void kernel_launch(void* const* d_in, const int* in_sizes, int n_in,
                              void* d_out, int out_size, void* d_ws,
                              size_t ws_size, hipStream_t stream) {
  (void)in_sizes; (void)n_in; (void)out_size; (void)ws_size;
  const float* X    = (const float*)d_in[0];
  const float* latW = (const float*)d_in[1];
  const float* latB = (const float*)d_in[2];
  const float* LP[3][16];
  for (int l = 0; l < 3; ++l)
    for (int j = 0; j < 16; ++j) LP[l][j] = (const float*)d_in[3 + 16 * l + j];
  const float* peW  = (const float*)d_in[51];
  const float* peb  = (const float*)d_in[52];
  const float* sW1  = (const float*)d_in[53];
  const float* sb1  = (const float*)d_in[54];
  const float* sW2  = (const float*)d_in[55];
  const float* sb2  = (const float*)d_in[56];
  const float* taW1 = (const float*)d_in[57];
  const float* tab1 = (const float*)d_in[58];
  const float* taW2 = (const float*)d_in[59];
  const float* tab2 = (const float*)d_in[60];
  const float* raW1 = (const float*)d_in[61];
  const float* rab1 = (const float*)d_in[62];
  const float* raW2 = (const float*)d_in[63];
  const float* rab2 = (const float*)d_in[64];
  float* OUT = (float*)d_out;

  char* base = (char*)d_ws;
  size_t cur = 0;
  auto alloc = [&](size_t bytes) -> void* {
    void* p = base + cur;
    cur += (bytes + 255) & ~(size_t)255;
    return p;
  };

  // bf16 transposed weights
  bf16 *WqT[3], *WkT[3], *WvT[3], *WoT[3], *W1T[3], *W2T[3];
  for (int l = 0; l < 3; ++l) {
    WqT[l] = (bf16*)alloc(64 * 64 * 2);
    WkT[l] = (bf16*)alloc(64 * 64 * 2);
    WvT[l] = (bf16*)alloc(64 * 64 * 2);
    WoT[l] = (bf16*)alloc(64 * 64 * 2);
    W1T[l] = (bf16*)alloc((size_t)DFF * 64 * 2);
    W2T[l] = (bf16*)alloc((size_t)64 * DFF * 2);
  }
  bf16* peWT  = (bf16*)alloc(128 * 32 * 2);
  bf16* sW1T  = (bf16*)alloc((size_t)SHARED * SHARED * 2);
  bf16* sW2T  = (bf16*)alloc((size_t)SHARED * SHARED * 2);
  bf16* taW1T = (bf16*)alloc((size_t)64 * SHARED * 2);
  bf16* raW1T = (bf16*)alloc((size_t)64 * SHARED * 2);
  bf16* taW2T = (bf16*)alloc(96 * 64 * 2);
  bf16* raW2T = (bf16*)alloc(96 * 64 * 2);

  // activations
  float* hF0   = (float*)alloc(ROWS * DD * 4);
  float* hF1   = (float*)alloc(ROWS * DD * 4);
  float* projF = (float*)alloc(ROWS * DD * 4);
  bf16* hB  = (bf16*)alloc(ROWS * DD * 2);
  bf16* qB  = (bf16*)alloc(ROWS * DD * 2);
  bf16* kB  = (bf16*)alloc(ROWS * DD * 2);
  bf16* vB  = (bf16*)alloc(ROWS * DD * 2);
  bf16* aoB = (bf16*)alloc(ROWS * DD * 2);
  bf16* comb = (bf16*)alloc((size_t)512 * SHARED * 2);
  bf16* u1   = (bf16*)alloc((size_t)512 * SHARED * 2);
  bf16* fB   = (bf16*)alloc((size_t)512 * SHARED * 2);
  bf16* gt = (bf16*)alloc(SS * 64 * 2);
  bf16* gr = (bf16*)alloc(SS * 64 * 2);
  float* tpF = (float*)alloc(SS * PL * 4);
  float* rpF = (float*)alloc(SS * PL * 4);
  float* trendF = (float*)alloc(ROWS * 4);
  float* residF = (float*)alloc(ROWS * 4);
  float* mfBuf  = (float*)alloc(ROWS * 4);

  auto cdiv = [](long a, long b) { return (a + b - 1) / b; };
  auto T = [&](const float* W, bf16* WT, int K, int N, int Kpad) {
    long tot = (long)N * Kpad;
    transpose_to_bf16<<<dim3((unsigned)cdiv(tot, 256)), dim3(256), 0, stream>>>(
        W, WT, K, N, Kpad);
  };
  // generic (odd shapes)
  auto GEMM = [&](const bf16* A, const bf16* BT, const float* bias, float* oF,
                  bf16* oB, long M, long N, long K, long lda, int relu) {
    long waves = (M / 16) * (N / 16);
    gemm_wmma<<<dim3((unsigned)cdiv(waves * 32, 128)), dim3(128), 0, stream>>>(
        A, BT, bias, oF, oB, (int)M, (int)N, (int)K, lda, relu, (int)waves);
  };
  // blocked 32x64 per wave (M%32==0, N%64==0)
  auto GEMMB = [&](const bf16* A, const bf16* BT, const float* bias, float* oF,
                   bf16* oB, long M, long N, long K, long lda, int relu) {
    long waves = (M / 32) * (N / 64);
    gemm_wmma_blk<<<dim3((unsigned)cdiv(waves * 32, 128)), dim3(128), 0,
                    stream>>>(A, BT, bias, oF, oB, (int)M, (int)N, (int)K, lda,
                              relu, (int)waves);
  };

  // ---- weight prep -------------------------------------------------------
  for (int l = 0; l < 3; ++l) {
    T(LP[l][0], WqT[l], 64, 64, 64);
    T(LP[l][2], WkT[l], 64, 64, 64);
    T(LP[l][4], WvT[l], 64, 64, 64);
    T(LP[l][6], WoT[l], 64, 64, 64);
    T(LP[l][10], W1T[l], 64, DFF, 64);
    T(LP[l][12], W2T[l], DFF, 64, DFF);
  }
  T(peW, peWT, 16, 128, 32);
  T(sW1, sW1T, SHARED, SHARED, SHARED);
  T(sW2, sW2T, SHARED, SHARED, SHARED);
  T(taW1, taW1T, SHARED, 64, SHARED);
  T(raW1, raW1T, SHARED, 64, SHARED);
  T(taW2, taW2T, 64, 96, 64);
  T(raW2, raW2T, 64, 96, 64);

  // ---- latent transformer ------------------------------------------------
  embed_latent<<<dim3((unsigned)cdiv(ROWS * DD, 256)), dim3(256), 0, stream>>>(
      X, latW, latB, hF0, hB);

  float* hcur = hF0;
  float* hnext = hF1;
  for (int l = 0; l < 3; ++l) {
    GEMMB(hB, WqT[l], LP[l][1], nullptr, qB, ROWS, DD, DD, DD, 0);
    GEMMB(hB, WkT[l], LP[l][3], nullptr, kB, ROWS, DD, DD, DD, 0);
    GEMMB(hB, WvT[l], LP[l][5], nullptr, vB, ROWS, DD, DD, DD, 0);
    attn_fused<<<dim3(SS * HH * 21 / 2), dim3(64), 0, stream>>>(qB, kB, vB, aoB);
    GEMMB(aoB, WoT[l], LP[l][7], projF, nullptr, ROWS, DD, DD, DD, 0);
    residual_ln<<<dim3((unsigned)cdiv(ROWS * 32, 256)), dim3(256), 0, stream>>>(
        hcur, projF, LP[l][8], LP[l][9], hnext, hB, ROWS);
    ffn_fused<<<dim3((unsigned)(ROWS / 16)), dim3(128), 0, stream>>>(
        hB, W1T[l], LP[l][11], W2T[l], LP[l][13], projF);
    residual_ln<<<dim3((unsigned)cdiv(ROWS * 32, 256)), dim3(256), 0, stream>>>(
        hnext, projF, LP[l][14], LP[l][15], hcur, hB, ROWS);
  }
  latent_mean<<<dim3(SS * DD / 256), dim3(256), 0, stream>>>(hcur, comb);

  // ---- Kalman trend/residual --------------------------------------------
  kalman_smooth<<<dim3(1), dim3(256), 0, stream>>>(X, trendF, residF, mfBuf);

  // ---- patch embedding into comb ----------------------------------------
  patch_embed<<<dim3(2 * 672 * 8 * 32 / 128), dim3(128), 0, stream>>>(
      trendF, residF, peWT, peb, comb);

  // ---- shared MLP + heads -----------------------------------------------
  GEMMB(comb, sW1T, sb1, nullptr, u1, 512, SHARED, SHARED, SHARED, 1);
  GEMMB(u1, sW2T, sb2, nullptr, fB, 512, SHARED, SHARED, SHARED, 0);
  GEMMB(fB, taW1T, tab1, nullptr, gt, SS, 64, SHARED, SHARED, 1);
  GEMMB(fB + (size_t)SS * SHARED, raW1T, rab1, nullptr, gr, SS, 64, SHARED,
        SHARED, 1);
  GEMM(gt, taW2T, tab2, tpF, nullptr, SS, PL, 64, 64, 0);
  GEMM(gr, raW2T, rab2, rpF, nullptr, SS, PL, 64, 64, 0);

  final_out<<<dim3(SS * PL / 256), dim3(256), 0, stream>>>(tpF, rpF, OUT);
}
